// GNN_71640054497345
// MI455X (gfx1250) — matrix operand order repaired and checked by reading, not compiled, not run on hardware
//
#include <hip/hip_runtime.h>

// ---------------------------------------------------------------------------
// GCN inference on MI455X (gfx1250, wave32).
//   - Dense transforms:   v_wmma_f32_16x16x32_bf16 (16x16 tile per wave)
//   - Sparse aggregation: float4 gather + global_atomic_add_f32 scatter
//   - L2 (192MB) holds hw/agg (<=25.6MB each) -> edge traffic is L2-resident.
// ---------------------------------------------------------------------------

typedef __attribute__((ext_vector_type(16))) __bf16 v16bf;
typedef __attribute__((ext_vector_type(8)))  float  v8f;

#define BN_EPS 1e-5f

// ---- degree / dinv --------------------------------------------------------
__global__ void k_init_deg(float* __restrict__ deg, int N) {
    int i = blockIdx.x * blockDim.x + threadIdx.x;
    if (i < N) deg[i] = 1.0f;   // self-loop
}

__global__ void k_count_deg(const int* __restrict__ dst, float* __restrict__ deg, int E) {
    int e = blockIdx.x * blockDim.x + threadIdx.x;
    if (e < E) atomicAdd(&deg[dst[e]], 1.0f);
}

__global__ void k_deg_to_dinv(float* __restrict__ deg, int N) {
    int i = blockIdx.x * blockDim.x + threadIdx.x;
    if (i < N) deg[i] = rsqrtf(deg[i]);
}

// ---- WMMA GEMM: Out[M,Nf] = A[M,K] @ W[K,Nf] ------------------------------
// One wave computes one 16x16 output tile. bf16 inputs, f32 accumulate.
// Per-lane layouts per CDNA5 ISA 7.12.2:
//   A (16x32 bf16): row M = lane&15; VGPR v packs K = (v>>2)*16 + (lane>>4)*8 + (v&3)*2 (+1)
//   B (32x16 bf16): col N = lane&15; VGPR v packs K = (v>>2)*16 + (lane>>4)*8 + (v&3)*2 (+1)
//   D (16x16 f32):  VGPR v holds row (v + 8*(lane>>4)), col = lane&15
__global__ __launch_bounds__(256) void k_gemm_wmma(
    const float* __restrict__ A, const float* __restrict__ W,
    float* __restrict__ Out, int M, int K, int Nf)
{
    const int lane  = threadIdx.x & 31;
    const int wave  = threadIdx.x >> 5;
    const int tilesN = Nf >> 4;
    const int gtile  = blockIdx.x * 8 + wave;
    const int totalTiles = (M >> 4) * tilesN;
    if (gtile >= totalTiles) return;          // wave-uniform: EXEC stays all-1s

    const int tileM = gtile / tilesN;
    const int tileN = gtile - tileM * tilesN;
    const int l15   = lane & 15;
    const int half  = lane >> 4;
    const int mrow  = (tileM << 4) + l15;
    const int ncol  = (tileN << 4) + l15;

    v8f acc = {0.f, 0.f, 0.f, 0.f, 0.f, 0.f, 0.f, 0.f};

    for (int k0 = 0; k0 < K; k0 += 32) {
        v16bf a, b;
#pragma unroll
        for (int v = 0; v < 8; ++v) {
            const int ka = k0 + ((v >> 2) << 4) + (half << 3) + ((v & 3) << 1);
            const float* ap = A + (size_t)mrow * K + ka;
            a[2 * v]     = (__bf16)ap[0];
            a[2 * v + 1] = (__bf16)ap[1];
            const int kb = k0 + ((v >> 2) << 4) + (half << 3) + ((v & 3) << 1);
            const float* bp = W + (size_t)kb * Nf + ncol;
            b[2 * v]     = (__bf16)bp[0];
            b[2 * v + 1] = (__bf16)bp[Nf];
        }
        acc = __builtin_amdgcn_wmma_f32_16x16x32_bf16(
            /*neg_a=*/false, a, /*neg_b=*/false, b,
            /*c_mod=*/(short)0, acc, /*reuse_a=*/false, /*reuse_b=*/false);
    }

#pragma unroll
    for (int v = 0; v < 8; ++v) {
        const int r = (tileM << 4) + (half << 3) + v;
        Out[(size_t)r * Nf + ncol] = acc[v];
    }
}

// ---- self-loop contribution: agg = hw * dinv^2 (initializes agg fully) ----
__global__ void k_self_agg(const float* __restrict__ hw, const float* __restrict__ dinv,
                           float* __restrict__ agg, int N, int F)
{
    int i = blockIdx.x * blockDim.x + threadIdx.x;
    if (i >= N * F) return;
    int n = i / F;
    float di = dinv[n];
    agg[i] = hw[i] * di * di;
}

// ---- edge scatter: agg[dst] += hw[src] * dinv[src]*dinv[dst] --------------
// 4 features per thread: float4 gather + 4 f32 global atomics.
__global__ void k_edge_agg(const float* __restrict__ hw,
                           const int* __restrict__ src, const int* __restrict__ dst,
                           const float* __restrict__ dinv, float* __restrict__ agg,
                           int E, int F)
{
    int t = blockIdx.x * blockDim.x + threadIdx.x;
    int chunks = F >> 2;
    if (t >= E * chunks) return;
    int e = t / chunks;
    int c = (t - e * chunks) << 2;
    int s = src[e];
    int d = dst[e];
    float coef = dinv[s] * dinv[d];
    const float4 v = *(const float4*)(hw + (size_t)s * F + c);
    float* o = agg + (size_t)d * F + c;
    atomicAdd(o + 0, v.x * coef);
    atomicAdd(o + 1, v.y * coef);
    atomicAdd(o + 2, v.z * coef);
    atomicAdd(o + 3, v.w * coef);
}

// ---- fused bias + BatchNorm(eval) + ReLU, in place ------------------------
__global__ void k_bn_relu(float* __restrict__ h, const float* __restrict__ bias,
                          const float* __restrict__ g, const float* __restrict__ be,
                          const float* __restrict__ m, const float* __restrict__ v,
                          int N, int F)
{
    int i = blockIdx.x * blockDim.x + threadIdx.x;
    if (i >= N * F) return;
    int f = i % F;
    float x = h[i] + bias[f];
    x = g[f] * (x - m[f]) * rsqrtf(v[f] + BN_EPS) + be[f];
    h[i] = fmaxf(x, 0.0f);
}

// ---- final FC 16->8 + sigmoid ---------------------------------------------
__global__ void k_fc_sigmoid(const float* __restrict__ h, const float* __restrict__ Wf,
                             const float* __restrict__ bf, float* __restrict__ out, int N)
{
    int i = blockIdx.x * blockDim.x + threadIdx.x;
    if (i >= N * 8) return;
    int n = i >> 3;
    int c = i & 7;
    const float* hp = h + (size_t)n * 16;
    float s = bf[c];
#pragma unroll
    for (int k = 0; k < 16; ++k) s += hp[k] * Wf[k * 8 + c];
    out[i] = 1.0f / (1.0f + __expf(-s));
}

// ---------------------------------------------------------------------------
extern "C" void kernel_launch(void* const* d_in, const int* in_sizes, int n_in,
                              void* d_out, int out_size, void* d_ws, size_t ws_size,
                              hipStream_t stream)
{
    const float* x    = (const float*)d_in[0];
    const int*   eidx = (const int*)  d_in[1];
    const float* W1  = (const float*)d_in[2];  const float* b1  = (const float*)d_in[3];
    const float* g1  = (const float*)d_in[4];  const float* be1 = (const float*)d_in[5];
    const float* m1  = (const float*)d_in[6];  const float* v1  = (const float*)d_in[7];
    const float* W2  = (const float*)d_in[8];  const float* b2  = (const float*)d_in[9];
    const float* g2  = (const float*)d_in[10]; const float* be2 = (const float*)d_in[11];
    const float* m2  = (const float*)d_in[12]; const float* v2  = (const float*)d_in[13];
    const float* W3  = (const float*)d_in[14]; const float* b3  = (const float*)d_in[15];
    const float* g3  = (const float*)d_in[16]; const float* be3 = (const float*)d_in[17];
    const float* m3  = (const float*)d_in[18]; const float* v3  = (const float*)d_in[19];
    const float* fcW = (const float*)d_in[20]; const float* fcb = (const float*)d_in[21];

    const int N = in_sizes[0] / 128;   // 100000 (divisible by 16)
    const int E = in_sizes[1] / 2;     // 1600000
    const int* src = eidx;
    const int* dst = eidx + E;

    // workspace: dinv (N f32) | bufA (N*64 f32) | bufB (N*64 f32)  ~= 51.6 MB
    char* ws = (char*)d_ws;
    size_t offA = ((size_t)N * 4 + 255) & ~(size_t)255;
    size_t sz64 = ((size_t)N * 64 * 4 + 255) & ~(size_t)255;
    float* dinv = (float*)ws;
    float* bufA = (float*)(ws + offA);
    float* bufB = (float*)(ws + offA + sz64);

    const dim3 blk(256);
    auto nblk = [](long n) { return dim3((unsigned)((n + 255) / 256)); };

    // degrees -> dinv (recomputed every call; deterministic up to atomic fp order)
    k_init_deg   <<<nblk(N), blk, 0, stream>>>(dinv, N);
    k_count_deg  <<<nblk(E), blk, 0, stream>>>(dst, dinv, E);
    k_deg_to_dinv<<<nblk(N), blk, 0, stream>>>(dinv, N);

    auto layer = [&](const float* h, int Fin, int Fout,
                     const float* W, const float* b, const float* g,
                     const float* be, const float* m, const float* v) {
        const long tiles = (long)(N >> 4) * (Fout >> 4);
        k_gemm_wmma<<<dim3((unsigned)((tiles + 7) / 8)), blk, 0, stream>>>(h, W, bufA, N, Fin, Fout);
        const long nf = (long)N * Fout;
        k_self_agg <<<nblk(nf), blk, 0, stream>>>(bufA, dinv, bufB, N, Fout);
        const long ework = (long)E * (Fout >> 2);
        k_edge_agg <<<nblk(ework), blk, 0, stream>>>(bufA, src, dst, dinv, bufB, E, Fout);
        k_bn_relu  <<<nblk(nf), blk, 0, stream>>>(bufB, b, g, be, m, v, N, Fout);
    };

    layer(x,    128, 64, W1, b1, g1, be1, m1, v1);   // h1 in bufB
    layer(bufB,  64, 32, W2, b2, g2, be2, m2, v2);   // h2 in bufB
    layer(bufB,  32, 16, W3, b3, g3, be3, m3, v3);   // h3 in bufB

    k_fc_sigmoid<<<nblk((long)N * 8), blk, 0, stream>>>(bufB, fcW, fcb, (float*)d_out, N);
}
